// ContentAddressableDNC_75144747811275
// MI455X (gfx1250) — matrix operand order repaired and checked by reading, not compiled, not run on hardware
//
#include <hip/hip_runtime.h>
#include <hip/hip_bf16.h>
#include <math.h>

// ---------------------------------------------------------------------------
// ContentAddressableDNC for MI455X (gfx1250, wave32, WMMA 16x16x32 f16)
//   B=16, T=512, D=64, H=128, S=100, Ms=20
// Pipeline:
//   K0 pack  : pack 8 head weight mats (128x20 f32) -> Wpk f16 (128x160) + bias
//   K1 lstm  : single workgroup (32 waves) persistent LSTM; per step one
//              (16x192)@(192x512) GEMM via 6 WMMA per wave, B-frags in VGPRs,
//              x_t staged via global_load_async_to_lds_b32 double buffer
//   K2 gemm  : ctrl (8192x128 f16) @ Wpk (128x160 f16) -> proj (8192x160 f32)
//   K3 heads : per-(b,t) block: softmax keys, neg-cos addressing, 2 reads,
//              2 sequential memory writes -> out + mem tensor
// ---------------------------------------------------------------------------

typedef __attribute__((ext_vector_type(16))) _Float16 v16h;
typedef __attribute__((ext_vector_type(8)))  float    v8f;

union HF16 { v16h v; _Float16 h[16]; };

#define BB 16
#define TT 512
#define DD 64
#define HH 128
#define SS 100
#define MS 20
#define KTOT 192            // D + H
#define NTOT 512            // 4*H
#define OUTC 168            // H + 2*Ms
#define NPRJ 160            // 8 heads * Ms

#if __has_builtin(__builtin_amdgcn_tanhf)
__device__ __forceinline__ float fast_tanh(float x) { return __builtin_amdgcn_tanhf(x); }
#else
__device__ __forceinline__ float fast_tanh(float x) {
  float e = __expf(2.0f * x);
  return (e - 1.0f) / (e + 1.0f);
}
#endif
__device__ __forceinline__ float sigmoidf_(float x) { return 0.5f * fast_tanh(0.5f * x) + 0.5f; }

// async copy of 4 bytes per lane: LDS[lds_addr] = MEM[sbase + voff]  (GVS mode)
__device__ __forceinline__ void async_ld_b32(unsigned lds_addr, unsigned long long sbase,
                                             unsigned voff) {
  asm volatile("global_load_async_to_lds_b32 %0, %1, %2"
               :: "v"(lds_addr), "v"(voff), "s"(sbase)
               : "memory");
}
__device__ __forceinline__ void wait_asynccnt0() {
  asm volatile("s_wait_asynccnt 0x0" ::: "memory");
}

// ---------------------------------------------------------------------------
// K0: pack head weights/biases: column blocks [rWk0 rWk1 wWk0 wWk1 wWe0 wWe1 wWa0 wWa1]
// ---------------------------------------------------------------------------
__global__ void dnc_pack_kernel(const float* __restrict__ rWk, const float* __restrict__ rbk,
                                const float* __restrict__ wWk, const float* __restrict__ wbk,
                                const float* __restrict__ wWe, const float* __restrict__ wbe,
                                const float* __restrict__ wWa, const float* __restrict__ wba,
                                _Float16* __restrict__ Wpk, float* __restrict__ biasPk) {
  int i = blockIdx.x * blockDim.x + threadIdx.x;
  if (i < HH * NPRJ) {
    int k = i / NPRJ, col = i % NPRJ;
    int blk = col / MS, m = col % MS;
    const float* src;
    switch (blk) {
      case 0: src = rWk;               break;
      case 1: src = rWk + HH * MS;     break;
      case 2: src = wWk;               break;
      case 3: src = wWk + HH * MS;     break;
      case 4: src = wWe;               break;
      case 5: src = wWe + HH * MS;     break;
      case 6: src = wWa;               break;
      default: src = wWa + HH * MS;    break;
    }
    Wpk[k * NPRJ + col] = (_Float16)src[k * MS + m];
  }
  if (i < NPRJ) {
    int blk = i / MS, m = i % MS;
    const float* s;
    switch (blk) {
      case 0: s = rbk;        break;
      case 1: s = rbk + MS;   break;
      case 2: s = wbk;        break;
      case 3: s = wbk + MS;   break;
      case 4: s = wbe;        break;
      case 5: s = wbe + MS;   break;
      case 6: s = wba;        break;
      default: s = wba + MS;  break;
    }
    biasPk[i] = s[m];
  }
}

// ---------------------------------------------------------------------------
// K1: persistent single-workgroup LSTM. 1024 threads = 32 waves.
// Wave w owns output columns [16w, 16w+16) of z (16x512).
// B-fragments (6 x v16h = 48 VGPRs/lane) hold [Wx;Wr] columns, loaded once.
// x_{t+1} prefetched via async-to-LDS while WMMA/gates of step t run.
// ---------------------------------------------------------------------------
__global__ void __launch_bounds__(1024)
dnc_lstm_kernel(const float* __restrict__ inputs,  // (B,T,D)
                const float* __restrict__ Wx,      // (D,4H)
                const float* __restrict__ Wr,      // (H,4H)
                const float* __restrict__ lb,      // (4H)
                _Float16* __restrict__ ctrlh,      // (B*T, H) f16
                float* __restrict__ out) {         // (B,T,OUTC)
  __shared__ _Float16 A_lds[BB * KTOT];   // [x_t | h]  row-major, 6 KB
  __shared__ float    z_lds[BB * NTOT];   // 32 KB
  __shared__ float    c_lds[BB * HH];     // 8 KB
  __shared__ float    xstg[2][BB * DD];   // async double buffer, 8 KB

  const int tid    = threadIdx.x;
  const int lane   = tid & 31;
  const int wv     = tid >> 5;            // 0..31 -> N tile
  const int laneLo = lane & 15;
  const int laneHi = lane >> 4;           // 0 or 1
  const int nCol   = (wv << 4) + laneLo;  // 0..511

  // ---- preload B fragments (combined [Wx;Wr], K=192) ----
  HF16 bfrag[6];
  #pragma unroll
  for (int f = 0; f < 6; ++f) {
    #pragma unroll
    for (int v = 0; v < 8; ++v) {
      int k0 = f * 32 + laneHi * 16 + 2 * v;   // even
      int k1 = k0 + 1;
      float b0 = (k0 < DD) ? Wx[k0 * NTOT + nCol] : Wr[(k0 - DD) * NTOT + nCol];
      float b1 = (k1 < DD) ? Wx[k1 * NTOT + nCol] : Wr[(k1 - DD) * NTOT + nCol];
      bfrag[f].h[2 * v]     = (_Float16)b0;
      bfrag[f].h[2 * v + 1] = (_Float16)b1;
    }
  }
  const float bias = lb[nCol];

  // ---- init c = 0, h-region of A = 0 ----
  for (int j = tid; j < BB * HH; j += 1024) c_lds[j] = 0.0f;
  for (int j = tid; j < BB * KTOT; j += 1024) A_lds[j] = (_Float16)0.0f;
  __syncthreads();

  // thread's private element of x: (b, d); same wave issues + consumes it,
  // so a per-wave s_wait_asynccnt(0) is sufficient (no barrier needed).
  const int xb = tid >> 6, xd = tid & 63;
  const unsigned xoff0 = (unsigned)((xb * TT * DD + xd) * 4);
  const unsigned long long ibase = (unsigned long long)(uintptr_t)inputs;

  // prologue: kick off x_0
  async_ld_b32((unsigned)(uintptr_t)&xstg[0][tid], ibase, xoff0);

  for (int t = 0; t < TT; ++t) {
    const int cur = t & 1;
    // consume x_t, then immediately prefetch x_{t+1} (overlaps WMMA + gates)
    wait_asynccnt0();
    A_lds[xb * KTOT + xd] = (_Float16)xstg[cur][tid];
    if (t + 1 < TT)
      async_ld_b32((unsigned)(uintptr_t)&xstg[1 - cur][tid], ibase,
                   xoff0 + (unsigned)((t + 1) * DD * 4));
    __syncthreads();

    // ---- z tile = [x|h] @ [Wx;Wr] ----
    v8f acc = {};
    #pragma unroll
    for (int f = 0; f < 6; ++f) {
      HF16 a;
      #pragma unroll
      for (int v = 0; v < 8; ++v) {
        int kl = laneHi * 16 + 2 * v;
        int k  = f * 32 + kl;
        a.h[2 * v]     = A_lds[laneLo * KTOT + k];
        a.h[2 * v + 1] = A_lds[laneLo * KTOT + k + 1];
      }
      acc = __builtin_amdgcn_wmma_f32_16x16x32_f16(
          false, a.v, false, bfrag[f].v, (short)0, acc, false, false);
    }
    // epilogue: D layout M = r + 8*laneHi, N = nCol
    #pragma unroll
    for (int r = 0; r < 8; ++r)
      z_lds[(laneHi * 8 + r) * NTOT + nCol] = acc[r] + bias;
    __syncthreads();

    // ---- gates: 2048 (b,hid) pairs, 2 per thread ----
    #pragma unroll
    for (int e = 0; e < 2; ++e) {
      int idx = tid + e * 1024;
      int b = idx >> 7, hid = idx & 127;
      float zi = z_lds[b * NTOT + hid];
      float zf = z_lds[b * NTOT + HH + hid];
      float zg = z_lds[b * NTOT + 2 * HH + hid];
      float zo = z_lds[b * NTOT + 3 * HH + hid];
      float ig = sigmoidf_(zi);
      float fg = sigmoidf_(zf);
      float gg = fast_tanh(zg);
      float og = sigmoidf_(zo);
      float c  = fg * c_lds[b * HH + hid] + ig * gg;
      c_lds[b * HH + hid] = c;
      float h = og * fast_tanh(c);
      A_lds[b * KTOT + DD + hid] = (_Float16)h;       // feeds next step
      ctrlh[(b * TT + t) * HH + hid] = (_Float16)h;   // feeds projection GEMM
      out[(size_t)(b * TT + t) * OUTC + hid] = h;     // ctrl part of output
    }
    __syncthreads();
  }
}

// ---------------------------------------------------------------------------
// K2: proj = ctrl(8192x128 f16) @ Wpk(128x160 f16) + biasPk.  8 waves/block,
// 1 wave = one 16x16 output tile; 5120 tiles total.
// ---------------------------------------------------------------------------
__global__ void __launch_bounds__(256)
dnc_proj_gemm_kernel(const _Float16* __restrict__ ctrlh,
                     const _Float16* __restrict__ Wpk,
                     const float* __restrict__ biasPk,
                     float* __restrict__ proj) {
  const int lane   = threadIdx.x & 31;
  const int wv     = threadIdx.x >> 5;
  const int tile   = blockIdx.x * 8 + wv;           // 0..5119
  const int mTile  = tile / (NPRJ / 16);
  const int nTile  = tile % (NPRJ / 16);
  const int laneLo = lane & 15;
  const int laneHi = lane >> 4;
  const int row    = mTile * 16 + laneLo;           // A row (b*T+t)
  const int n      = nTile * 16 + laneLo;           // output col

  v8f acc = {};
  #pragma unroll
  for (int f = 0; f < 4; ++f) {                     // K = 128 = 4*32
    HF16 a, b;
    #pragma unroll
    for (int v = 0; v < 8; ++v) {
      int kl = laneHi * 16 + 2 * v;
      int ka = f * 32 + kl;
      a.h[2 * v]     = ctrlh[row * HH + ka];
      a.h[2 * v + 1] = ctrlh[row * HH + ka + 1];
      int kb = f * 32 + laneHi * 16 + 2 * v;
      b.h[2 * v]     = Wpk[kb * NPRJ + n];
      b.h[2 * v + 1] = Wpk[(kb + 1) * NPRJ + n];
    }
    acc = __builtin_amdgcn_wmma_f32_16x16x32_f16(
        false, a.v, false, b.v, (short)0, acc, false, false);
  }
  float bias = biasPk[n];
  #pragma unroll
  for (int r = 0; r < 8; ++r)
    proj[(size_t)(mTile * 16 + laneHi * 8 + r) * NPRJ + n] = acc[r] + bias;
}

// ---------------------------------------------------------------------------
// K3: one block (128 threads) per (b,t): reads + 2 sequential memory writes.
// proj row layout: [rk0 rk1 wk0 wk1 we0 we1 wa0 wa1] * 20
// Softmaxes done cooperatively by the first wave (wave32 shfl reductions).
// ---------------------------------------------------------------------------
__global__ void __launch_bounds__(128)
dnc_heads_kernel(const float* __restrict__ memory,   // (S,Ms)
                 const float* __restrict__ proj,     // (B*T, 160)
                 float* __restrict__ out) {          // out + mem tensor
  __shared__ float mem0[SS * MS];
  __shared__ float mem1[SS * MS];
  __shared__ float prow[NPRJ];
  __shared__ float kvec[MS];
  __shared__ float evec[MS];
  __shared__ float avec[MS];
  __shared__ float simw[SS];
  __shared__ float invn[SS];

  const int tid = threadIdx.x;
  const size_t bt = blockIdx.x;                 // == b*T + t
  const size_t memOutBase = (size_t)BB * TT * OUTC;

  __builtin_prefetch(memory + tid * 16, 0, 0);  // global_prefetch_b8

  for (int j = tid; j < SS * MS; j += 128) mem0[j] = memory[j];
  if (tid < NPRJ) prow[tid] = proj[bt * NPRJ + tid];
  __syncthreads();

  // inverse L2 norms of original memory rows
  if (tid < SS) {
    float s = 0.0f;
    #pragma unroll
    for (int m = 0; m < MS; ++m) { float v = mem0[tid * MS + m]; s += v * v; }
    invn[tid] = rsqrtf(fmaxf(s, 1e-12f));
  }
  __syncthreads();

  // ------- helper lambdas (first-wave cooperative) -------
  auto key_softmax_l2n = [&](int base) {
    if (tid < 32) {
      float v = (tid < MS) ? prow[base + tid] : -1e30f;
      float mx = v;
      #pragma unroll
      for (int o = 16; o; o >>= 1) mx = fmaxf(mx, __shfl_xor(mx, o, 32));
      float e = (tid < MS) ? __expf(v - mx) : 0.0f;
      float sum = e;
      #pragma unroll
      for (int o = 16; o; o >>= 1) sum += __shfl_xor(sum, o, 32);
      float kv = e / sum;
      float nrm = kv * kv;
      #pragma unroll
      for (int o = 16; o; o >>= 1) nrm += __shfl_xor(nrm, o, 32);
      float rn = rsqrtf(fmaxf(nrm, 1e-12f));
      if (tid < MS) kvec[tid] = kv * rn;
    }
  };
  auto softmax100 = [&]() {
    if (tid < 32) {
      float mx = -1e30f;
      for (int s = tid; s < SS; s += 32) mx = fmaxf(mx, simw[s]);
      #pragma unroll
      for (int o = 16; o; o >>= 1) mx = fmaxf(mx, __shfl_xor(mx, o, 32));
      float sum = 0.0f;
      for (int s = tid; s < SS; s += 32) { float e = __expf(simw[s] - mx); simw[s] = e; sum += e; }
      #pragma unroll
      for (int o = 16; o; o >>= 1) sum += __shfl_xor(sum, o, 32);
      float inv = 1.0f / sum;
      for (int s = tid; s < SS; s += 32) simw[s] *= inv;
    }
  };

  // ---------------- read heads (use original memory) ----------------
  for (int head = 0; head < 2; ++head) {
    key_softmax_l2n(head * MS);
    __syncthreads();
    if (tid < SS) {
      float d = 0.0f;
      #pragma unroll
      for (int m = 0; m < MS; ++m) d += kvec[m] * mem0[tid * MS + m];
      simw[tid] = -d * invn[tid];
    }
    __syncthreads();
    softmax100();
    __syncthreads();
    if (tid < MS) {
      float r = 0.0f;
      for (int s = 0; s < SS; ++s) r += simw[s] * mem0[s * MS + tid];
      out[bt * OUTC + HH + head * MS + tid] = r;
    }
    __syncthreads();
  }

  // ---------------- write heads (sequential) ----------------
  for (int head = 0; head < 2; ++head) {
    const float* msrc = (head == 0) ? mem0 : mem1;
    key_softmax_l2n((2 + head) * MS);
    if (tid < MS) {
      evec[tid] = sigmoidf_(prow[(4 + head) * MS + tid]);
      avec[tid] = prow[(6 + head) * MS + tid];
    }
    __syncthreads();
    if (head == 1) {   // re-norm rows of mem1
      if (tid < SS) {
        float s = 0.0f;
        #pragma unroll
        for (int m = 0; m < MS; ++m) { float v = mem1[tid * MS + m]; s += v * v; }
        invn[tid] = rsqrtf(fmaxf(s, 1e-12f));
      }
      __syncthreads();
    }
    if (tid < SS) {
      float d = 0.0f;
      #pragma unroll
      for (int m = 0; m < MS; ++m) d += kvec[m] * msrc[tid * MS + m];
      simw[tid] = -d * invn[tid];
    }
    __syncthreads();
    softmax100();
    __syncthreads();
    // apply write
    if (head == 0) {
      for (int j = tid; j < SS * MS; j += 128) {
        int s = j / MS, m = j % MS;
        mem1[j] = (1.0f - simw[s] * evec[m]) * mem0[j] + simw[s] * avec[m];
      }
    } else {
      float* dst = out + memOutBase + bt * (SS * MS);
      for (int j = tid; j < SS * MS; j += 128) {
        int s = j / MS, m = j % MS;
        dst[j] = (1.0f - simw[s] * evec[m]) * mem1[j] + simw[s] * avec[m];
      }
    }
    __syncthreads();
  }
}

// ---------------------------------------------------------------------------
extern "C" void kernel_launch(void* const* d_in, const int* in_sizes, int n_in,
                              void* d_out, int out_size, void* d_ws, size_t ws_size,
                              hipStream_t stream) {
  const float* inputs = (const float*)d_in[0];
  const float* memory = (const float*)d_in[1];
  const float* Wx     = (const float*)d_in[2];
  const float* Wr     = (const float*)d_in[3];
  const float* lb     = (const float*)d_in[4];
  const float* rWk    = (const float*)d_in[5];
  const float* rbk    = (const float*)d_in[6];
  const float* wWk    = (const float*)d_in[7];
  const float* wbk    = (const float*)d_in[8];
  const float* wWe    = (const float*)d_in[9];
  const float* wbe    = (const float*)d_in[10];
  const float* wWa    = (const float*)d_in[11];
  const float* wba    = (const float*)d_in[12];
  float* out = (float*)d_out;

  char* ws = (char*)d_ws;
  // workspace layout (all 128B-aligned starts)
  _Float16* Wpk    = (_Float16*)(ws + 0);                      // 128*160*2   = 40960
  float*    biasPk = (float*)(ws + 41088);                     // 160*4       = 640
  _Float16* ctrlh  = (_Float16*)(ws + 41856);                  // 8192*128*2  = 2097152
  float*    proj   = (float*)(ws + 41856 + 2097152);           // 8192*160*4  = 5242880

  dnc_pack_kernel<<<81, 256, 0, stream>>>(rWk, rbk, wWk, wbk, wWe, wbe, wWa, wba, Wpk, biasPk);
  dnc_lstm_kernel<<<1, 1024, 0, stream>>>(inputs, Wx, Wr, lb, ctrlh, out);
  dnc_proj_gemm_kernel<<<640, 256, 0, stream>>>(ctrlh, Wpk, biasPk, proj);
  dnc_heads_kernel<<<BB * TT, 128, 0, stream>>>(memory, proj, out);
}
